// Graphormer3DEncoderLayer_49435073577656
// MI455X (gfx1250) — matrix operand order repaired and checked by reading, not compiled, not run on hardware
//
#include <hip/hip_runtime.h>
#include <hip/hip_bf16.h>
#include <math.h>

// ---------- types ----------
typedef __bf16 bf16;
typedef __attribute__((ext_vector_type(16))) __bf16 bf16x16;
typedef __attribute__((ext_vector_type(8)))  __bf16 bf16x8;
typedef __attribute__((ext_vector_type(8)))  float  f32x8;

union FragBf { bf16x16 v; bf16x8 h[2]; };

__device__ __forceinline__ f32x8 wmma_bf16(bf16x16 a, bf16x16 b, f32x8 c) {
  // D = A(16x32) * B(32x16) + C, f32 accum
  return __builtin_amdgcn_wmma_f32_16x16x32_bf16(
      /*neg_a=*/false, a, /*neg_b=*/false, b,
      /*c_mod=*/(short)0, c, /*reuse_a=*/false, /*reuse_b=*/false);
}

// ---------- CDNA5 async global->LDS copy (ASYNCcnt-tracked) via inline asm ----
// CDNA5 ISA: GLOBAL_LOAD_ASYNC_TO_LDS_B128, GV mode:
//   vdst = per-lane LDS byte address, vaddr = 64-bit global address, saddr=off.
// LDS byte offset = low 32 bits of a generic pointer into __shared__ space.
#if defined(__gfx1250__)
#define USE_ASYNC_ASM 1
#else
#define USE_ASYNC_ASM 0
#endif

__device__ __forceinline__ void copy16_g2l(const bf16* __restrict__ g, bf16* l) {
#if USE_ASYNC_ASM
  unsigned lds_off = (unsigned)(unsigned long long)l;  // low 32 bits = LDS offset
  unsigned long long gaddr = (unsigned long long)g;
  asm volatile("global_load_async_to_lds_b128 %0, %1, off"
               :: "v"(lds_off), "v"(gaddr)
               : "memory");
#else
  *(bf16x8*)l = *(const bf16x8*)g;
#endif
}

__device__ __forceinline__ void wait_async_copies() {
#if USE_ASYNC_ASM
  asm volatile("s_wait_asynccnt 0" ::: "memory");
#endif
}

// ---------- problem constants ----------
#define NNODE 512
#define NGRAPH 16
#define DMODEL 768
#define NHEAD 8
#define HD 96
#define NB (NGRAPH*NHEAD)   // 128 batched heads
#define ROWS (NNODE*NGRAPH) // 8192
#define FFN 3072
#define QSCALE 0.1020620726159658f  // 96^-0.5

// ======================================================================
// fp32 -> bf16 elementwise convert (for weights)
// ======================================================================
__global__ __launch_bounds__(256) void cvt_bf16_kernel(const float* __restrict__ in,
                                                       bf16* __restrict__ out, int n) {
  int i = blockIdx.x * 256 + threadIdx.x;
  if (i < n) out[i] = (bf16)in[i];
}

// ======================================================================
// LayerNorm over D=768, output bf16. One block (256 thr) per row.
// ======================================================================
__global__ __launch_bounds__(256) void ln_bf16_kernel(const float* __restrict__ x,
                                                      const float* __restrict__ g,
                                                      const float* __restrict__ b,
                                                      bf16* __restrict__ out) {
  int row = blockIdx.x;
  int tid = threadIdx.x;
  const float* xr = x + (size_t)row * DMODEL;
  float v0 = xr[tid], v1 = xr[tid + 256], v2 = xr[tid + 512];
  float s = v0 + v1 + v2;
  float ss = v0 * v0 + v1 * v1 + v2 * v2;
#pragma unroll
  for (int off = 16; off >= 1; off >>= 1) {
    s  += __shfl_xor(s,  off, 32);
    ss += __shfl_xor(ss, off, 32);
  }
  __shared__ float rs[8], rss[8];
  int lane = tid & 31, wave = tid >> 5;
  if (lane == 0) { rs[wave] = s; rss[wave] = ss; }
  __syncthreads();
  __shared__ float smu, srs;
  if (tid == 0) {
    float S = 0.f, SS = 0.f;
#pragma unroll
    for (int i = 0; i < 8; ++i) { S += rs[i]; SS += rss[i]; }
    float mu = S * (1.0f / DMODEL);
    float var = SS * (1.0f / DMODEL) - mu * mu;
    smu = mu; srs = rsqrtf(var + 1e-5f);
  }
  __syncthreads();
  float mu = smu, r = srs;
  bf16* o = out + (size_t)row * DMODEL;
  o[tid]       = (bf16)((v0 - mu) * r * g[tid]       + b[tid]);
  o[tid + 256] = (bf16)((v1 - mu) * r * g[tid + 256] + b[tid + 256]);
  o[tid + 512] = (bf16)((v2 - mu) * r * g[tid + 512] + b[tid + 512]);
}

// ======================================================================
// Tiled WMMA GEMM: C[M,N] = A[M,K](bf16, row-major) * W[N,K]^T (bf16) + bias
// Block tile 128x128, 8 waves (4x2), each wave 32x64 (2x4 WMMA tiles).
// EPI 0: store bf16           EPI 1: exact GELU, store bf16
// EPI 2: mask + residual add, store f32 (ld = N)
// EPI 3: QKV scatter into q_pk[b][n][hd], k_pk[b][n][hd], vT[b][hd][n]
// ======================================================================
#define BKP 40  // padded K stride in LDS (16B aligned, 20-bank rotation)

template <int EPI>
__global__ __launch_bounds__(256) void gemm_bf16_kernel(
    const bf16* __restrict__ A, const bf16* __restrict__ W,
    const float* __restrict__ bias, int M, int N, int K,
    const float* __restrict__ resIn, float* __restrict__ outF,
    bf16* __restrict__ outBf, const unsigned char* __restrict__ mask,
    bf16* __restrict__ qpk, bf16* __restrict__ kpk, bf16* __restrict__ vT) {
  __shared__ bf16 As[128 * BKP];
  __shared__ bf16 Bs[128 * BKP];
  const int tid = threadIdx.x;
  const int lane = tid & 31, wave = tid >> 5;
  const int wm = wave & 3, wn = wave >> 2;      // 4 x 2 wave grid
  const int half = lane >> 4, ln = lane & 15;
  const int m0 = blockIdx.y * 128, n0 = blockIdx.x * 128;

  f32x8 acc[2][4];
#pragma unroll
  for (int i = 0; i < 2; ++i)
#pragma unroll
    for (int j = 0; j < 4; ++j)
#pragma unroll
      for (int v = 0; v < 8; ++v) acc[i][j][v] = 0.f;

  for (int k0 = 0; k0 < K; k0 += 32) {
    __syncthreads();
    // A tile: 128 rows x 32 k -> 512 x bf16x8 vectors, 2 per thread
    // B tile: 128 rows (n) x 32 k -> 512 vectors, 2 per thread
#pragma unroll
    for (int it = 0; it < 2; ++it) {
      int vec = tid + it * 256;
      int row = vec >> 2, kk = (vec & 3) << 3;
      const bf16* asrc = &A[(size_t)(m0 + row) * K + k0 + kk];
      const bf16* bsrc = &W[(size_t)(n0 + row) * K + k0 + kk];
      copy16_g2l(asrc, &As[row * BKP + kk]);
      copy16_g2l(bsrc, &Bs[row * BKP + kk]);
      if (k0 + 32 < K) {
        __builtin_prefetch(asrc + 32, 0, 0);
        __builtin_prefetch(bsrc + 32, 0, 0);
      }
    }
    wait_async_copies();
    __syncthreads();

    FragBf af[2], bfm[4];
#pragma unroll
    for (int i = 0; i < 2; ++i) {  // A frag: lane=row, half picks K chunks {h*8.., 16+h*8..}
      const bf16* ar = &As[(wm * 32 + i * 16 + ln) * BKP];
      af[i].h[0] = *(const bf16x8*)(ar + half * 8);
      af[i].h[1] = *(const bf16x8*)(ar + 16 + half * 8);
    }
#pragma unroll
    for (int j = 0; j < 4; ++j) {  // B frag: lane=col, half picks K chunks {h*8.., 16+h*8..}
      const bf16* br = &Bs[(wn * 64 + j * 16 + ln) * BKP];
      bfm[j].h[0] = *(const bf16x8*)(br + half * 8);
      bfm[j].h[1] = *(const bf16x8*)(br + 16 + half * 8);
    }
#pragma unroll
    for (int i = 0; i < 2; ++i)
#pragma unroll
      for (int j = 0; j < 4; ++j)
        acc[i][j] = wmma_bf16(af[i].v, bfm[j].v, acc[i][j]);
  }

  // epilogue; C layout: element (lane, v) -> row m = v + 8*half, col n = lane&15
#pragma unroll
  for (int i = 0; i < 2; ++i)
#pragma unroll
    for (int j = 0; j < 4; ++j) {
      int col = n0 + wn * 64 + j * 16 + ln;
      float bv = bias[col];
#pragma unroll
      for (int v = 0; v < 8; ++v) {
        int row = m0 + wm * 32 + i * 16 + half * 8 + v;
        float val = acc[i][j][v] + bv;
        if (EPI == 0) {
          outBf[(size_t)row * N + col] = (bf16)val;
        } else if (EPI == 1) {
          float t = val;
          val = 0.5f * t * (1.0f + erff(t * 0.70710678118654752f));
          outBf[(size_t)row * N + col] = (bf16)val;
        } else if (EPI == 2) {
          if (mask) {
            int gg = row & 15, nn = row >> 4;
            if (!mask[gg * NNODE + nn]) val = 0.f;
          }
          outF[(size_t)row * N + col] = resIn[(size_t)row * N + col] + val;
        } else {  // EPI == 3: QKV scatter
          int which = col / DMODEL;
          int dd = col - which * DMODEL;
          int h = dd / HD, hd = dd - h * HD;
          int nn = row >> 4, gg = row & 15;
          int bh = gg * NHEAD + h;
          if (which == 0)
            qpk[((size_t)bh * NNODE + nn) * HD + hd] = (bf16)(val * QSCALE);
          else if (which == 1)
            kpk[((size_t)bh * NNODE + nn) * HD + hd] = (bf16)val;
          else
            vT[((size_t)bh * HD + hd) * NNODE + nn] = (bf16)val;
        }
      }
    }
}

// ======================================================================
// Flash attention. Grid (128 heads, 8 row-tiles), 128 thr = 4 waves.
// Each wave owns 16 rows; K=96 -> 3 WMMAs for QK^T; online softmax over
// 32-col steps; P staged via wave-private LDS into A-frag layout; PV = 6 WMMAs.
// ======================================================================
#define PPAD 40
__global__ __launch_bounds__(128) void attn_kernel(
    const bf16* __restrict__ qpk, const bf16* __restrict__ kpk,
    const bf16* __restrict__ vT, const float* __restrict__ attn_bias,
    bf16* __restrict__ opk) {
  __shared__ bf16 Psh[4][16 * PPAD];
  const int b = blockIdx.x;           // head-batch 0..127
  const int rt = blockIdx.y;          // row tile of 64
  const int lane = threadIdx.x & 31, wave = threadIdx.x >> 5;
  const int half = lane >> 4, ln = lane & 15;
  const int rBase = rt * 64 + wave * 16;
  bf16* Pw = Psh[wave];

  // Q fragments (rows rBase..rBase+15, K=96 -> 3 frags), kept in VGPRs
  FragBf qf[3];
  const bf16* qrow = qpk + ((size_t)b * NNODE + rBase + ln) * HD;
#pragma unroll
  for (int f = 0; f < 3; ++f) {
    qf[f].h[0] = *(const bf16x8*)(qrow + f * 32 + half * 8);
    qf[f].h[1] = *(const bf16x8*)(qrow + f * 32 + 16 + half * 8);
  }

  f32x8 o[6];
#pragma unroll
  for (int f = 0; f < 6; ++f)
#pragma unroll
    for (int v = 0; v < 8; ++v) o[f][v] = 0.f;
  float mi[8], li[8];
#pragma unroll
  for (int v = 0; v < 8; ++v) { mi[v] = -INFINITY; li[v] = 0.f; }

  for (int ct = 0; ct < NNODE / 32; ++ct) {
    float xs[2][8];
#pragma unroll
    for (int t = 0; t < 2; ++t) {
      int c0 = ct * 32 + t * 16;
      f32x8 s = {0.f, 0.f, 0.f, 0.f, 0.f, 0.f, 0.f, 0.f};
      const bf16* krow = kpk + ((size_t)b * NNODE + c0 + ln) * HD;
#pragma unroll
      for (int f = 0; f < 3; ++f) {
        FragBf kf;
        kf.h[0] = *(const bf16x8*)(krow + f * 32 + half * 8);
        kf.h[1] = *(const bf16x8*)(krow + f * 32 + 16 + half * 8);
        s = wmma_bf16(qf[f].v, kf.v, s);
      }
      const float* brow = attn_bias +
          (((size_t)b * NNODE + rBase + half * 8) * NNODE) + c0 + ln;
#pragma unroll
      for (int v = 0; v < 8; ++v) xs[t][v] = s[v] + brow[(size_t)v * NNODE];
    }
    // online softmax per row (16-lane group reductions)
#pragma unroll
    for (int v = 0; v < 8; ++v) {
      float tmax = fmaxf(xs[0][v], xs[1][v]);
#pragma unroll
      for (int off = 8; off >= 1; off >>= 1)
        tmax = fmaxf(tmax, __shfl_xor(tmax, off, 32));
      float mnew = fmaxf(mi[v], tmax);
      float alpha = __expf(mi[v] - mnew);
      float p0 = __expf(xs[0][v] - mnew);
      float p1 = __expf(xs[1][v] - mnew);
      float tsum = p0 + p1;
#pragma unroll
      for (int off = 8; off >= 1; off >>= 1) tsum += __shfl_xor(tsum, off, 32);
      li[v] = li[v] * alpha + tsum;
      mi[v] = mnew;
#pragma unroll
      for (int f = 0; f < 6; ++f) o[f][v] *= alpha;
      int m = half * 8 + v;
      Pw[m * PPAD + ln] = (bf16)p0;
      Pw[m * PPAD + 16 + ln] = (bf16)p1;
    }
    // P -> A-fragment (16x32), wave-private LDS (in-order DS, compiler waits)
    FragBf pa;
    {
      const bf16* pr = &Pw[ln * PPAD];
      pa.h[0] = *(const bf16x8*)(pr + half * 8);
      pa.h[1] = *(const bf16x8*)(pr + 16 + half * 8);
    }
    // PV: V^T[b][hd][node] -> contiguous K per lane
#pragma unroll
    for (int f = 0; f < 6; ++f) {
      FragBf vf;
      const bf16* vrow = vT + ((size_t)b * HD + f * 16 + ln) * NNODE + ct * 32;
      vf.h[0] = *(const bf16x8*)(vrow + half * 8);
      vf.h[1] = *(const bf16x8*)(vrow + 16 + half * 8);
      o[f] = wmma_bf16(pa.v, vf.v, o[f]);
    }
  }
  // finalize: divide by row sums, store head-major bf16
#pragma unroll
  for (int f = 0; f < 6; ++f)
#pragma unroll
    for (int v = 0; v < 8; ++v) {
      float val = o[f][v] / li[v];
      opk[((size_t)b * NNODE + rBase + half * 8 + v) * HD + f * 16 + ln] = (bf16)val;
    }
}

// ======================================================================
// Repack attention output [128][512][96] -> row-major [8192][768]
// ======================================================================
__global__ __launch_bounds__(256) void repack_attn_kernel(const bf16* __restrict__ apk,
                                                          bf16* __restrict__ arm) {
  size_t i = (size_t)blockIdx.x * 256 + threadIdx.x;  // dest index
  int row = (int)(i / DMODEL), col = (int)(i - (size_t)row * DMODEL);
  int nn = row >> 4, gg = row & 15;
  int h = col / HD, hd = col - h * HD;
  arm[i] = apk[((size_t)(gg * NHEAD + h) * NNODE + nn) * HD + hd];
}

// ======================================================================
// host-side launcher
// ======================================================================
extern "C" void kernel_launch(void* const* d_in, const int* in_sizes, int n_in,
                              void* d_out, int out_size, void* d_ws, size_t ws_size,
                              hipStream_t stream) {
  (void)in_sizes; (void)n_in; (void)out_size; (void)ws_size;
  const float* x        = (const float*)d_in[0];
  const float* abias    = (const float*)d_in[1];
  const unsigned char* mask = (const unsigned char*)d_in[2];
  const float* in_w     = (const float*)d_in[3];
  const float* in_b     = (const float*)d_in[4];
  const float* out_w    = (const float*)d_in[5];
  const float* out_b    = (const float*)d_in[6];
  const float* ln1_g    = (const float*)d_in[7];
  const float* ln1_b    = (const float*)d_in[8];
  const float* fc1_w    = (const float*)d_in[9];
  const float* fc1_b    = (const float*)d_in[10];
  const float* fc2_w    = (const float*)d_in[11];
  const float* fc2_b    = (const float*)d_in[12];
  const float* ln2_g    = (const float*)d_in[13];
  const float* ln2_b    = (const float*)d_in[14];
  float* out = (float*)d_out;

  size_t off = 0;
  char* base = (char*)d_ws;
  auto take = [&](size_t bytes) -> void* {
    void* p = base + off;
    off += (bytes + 255) & ~(size_t)255;
    return p;
  };
  const size_t HSZ = (size_t)NB * NNODE * HD * 2;  // 12.6 MB per head-tensor
  bf16* inw_bf  = (bf16*)take((size_t)3 * DMODEL * DMODEL * 2);
  bf16* outw_bf = (bf16*)take((size_t)DMODEL * DMODEL * 2);
  bf16* fc1w_bf = (bf16*)take((size_t)FFN * DMODEL * 2);
  bf16* fc2w_bf = (bf16*)take((size_t)DMODEL * FFN * 2);
  bf16* h_bf    = (bf16*)take((size_t)ROWS * DMODEL * 2);
  bf16* qpk     = (bf16*)take(HSZ);
  bf16* kpk     = (bf16*)take(HSZ);
  bf16* vT      = (bf16*)take(HSZ);
  bf16* apk     = (bf16*)take(HSZ);
  bf16* arm     = (bf16*)take((size_t)ROWS * DMODEL * 2);
  bf16* ffn_bf  = (bf16*)take((size_t)ROWS * FFN * 2);

  // 1. weights -> bf16
  cvt_bf16_kernel<<<(3 * DMODEL * DMODEL + 255) / 256, 256, 0, stream>>>(in_w, inw_bf, 3 * DMODEL * DMODEL);
  cvt_bf16_kernel<<<(DMODEL * DMODEL + 255) / 256, 256, 0, stream>>>(out_w, outw_bf, DMODEL * DMODEL);
  cvt_bf16_kernel<<<(FFN * DMODEL + 255) / 256, 256, 0, stream>>>(fc1_w, fc1w_bf, FFN * DMODEL);
  cvt_bf16_kernel<<<(DMODEL * FFN + 255) / 256, 256, 0, stream>>>(fc2_w, fc2w_bf, DMODEL * FFN);

  // 2. LN1 -> bf16
  ln_bf16_kernel<<<ROWS, 256, 0, stream>>>(x, ln1_g, ln1_b, h_bf);

  // 3. QKV projection with head-packing scatter epilogue
  gemm_bf16_kernel<3><<<dim3((3 * DMODEL) / 128, ROWS / 128), 256, 0, stream>>>(
      h_bf, inw_bf, in_b, ROWS, 3 * DMODEL, DMODEL,
      nullptr, nullptr, nullptr, nullptr, qpk, kpk, vT);

  // 4. flash attention (bias fused, softmax online)
  attn_kernel<<<dim3(NB, NNODE / 64), 128, 0, stream>>>(qpk, kpk, vT, abias, apk);

  // 5. repack heads -> row-major
  repack_attn_kernel<<<(ROWS * DMODEL) / 256, 256, 0, stream>>>(apk, arm);

  // 6. out-proj + padding mask + residual -> d_out (= x1, fp32)
  gemm_bf16_kernel<2><<<dim3(DMODEL / 128, ROWS / 128), 256, 0, stream>>>(
      arm, outw_bf, out_b, ROWS, DMODEL, DMODEL,
      x, out, nullptr, mask, nullptr, nullptr, nullptr);

  // 7. LN2 -> bf16
  ln_bf16_kernel<<<ROWS, 256, 0, stream>>>(out, ln2_g, ln2_b, h_bf);

  // 8. fc1 + exact GELU -> bf16
  gemm_bf16_kernel<1><<<dim3(FFN / 128, ROWS / 128), 256, 0, stream>>>(
      h_bf, fc1w_bf, fc1_b, ROWS, FFN, DMODEL,
      nullptr, nullptr, ffn_bf, nullptr, nullptr, nullptr, nullptr);

  // 9. fc2 + residual (in-place per-element RMW on d_out)
  gemm_bf16_kernel<2><<<dim3(DMODEL / 128, ROWS / 128), 256, 0, stream>>>(
      ffn_bf, fc2w_bf, fc2_b, ROWS, DMODEL, FFN,
      out, out, nullptr, nullptr, nullptr, nullptr, nullptr);
}
